// MambaIRUNet_68942815035929
// MI455X (gfx1250) — matrix operand order repaired and checked by reading, not compiled, not run on hardware
//
#include <hip/hip_runtime.h>
#include <math.h>

// ---------------------------------------------------------------------------
// MambaIR 3D selective-scan block for MI455X (gfx1250, wave32, WMMA).
//
// Shapes: B=1, H=W=Dd=16 -> L=4096, C=Din=48, N=16, R=3, K=8 directions.
// All GEMM operands are staged zero-padded to K=64 so WMMA fragment loads are
// branch-free, vectorized (float4 / global_load_b128) and need no masking.
// Scan: one wave per (k,d) channel, 32-lane chunked associative scan.
// ---------------------------------------------------------------------------

typedef __attribute__((ext_vector_type(16))) _Float16 v16h;
typedef __attribute__((ext_vector_type(8)))  float    v8f;

#define L_TOT   4096
#define DIN     48
#define KPAD    64
#define NSTATE  16
#define KDIR    8
#define CROWS   35   // R + 2N = 3 + 32

// ----------------------- WMMA fragment helpers (branch-free) ---------------
// A: 16x32 f16 tile of row-major Mx64 matrix (lda = 64, zero padded).
// Contiguous k order used here (not ISA table order): lane m = lane&15; lanes 0-15 hold K k0..k0+15, lanes 16-31
// hold K k0+16..k0+31, half j -> k = k0 + hi*16 + j.
__device__ __forceinline__ v16h load_a64(const float* __restrict__ A,
                                         int m0, int k0, int lane) {
  const int mi = lane & 15, hi = lane >> 4;
  const float* p = A + (size_t)(m0 + mi) * KPAD + k0 + hi * 16;
  const float4 r0 = ((const float4*)p)[0];
  const float4 r1 = ((const float4*)p)[1];
  const float4 r2 = ((const float4*)p)[2];
  const float4 r3 = ((const float4*)p)[3];
  v16h a;
  a[0]  = (_Float16)r0.x; a[1]  = (_Float16)r0.y; a[2]  = (_Float16)r0.z; a[3]  = (_Float16)r0.w;
  a[4]  = (_Float16)r1.x; a[5]  = (_Float16)r1.y; a[6]  = (_Float16)r1.z; a[7]  = (_Float16)r1.w;
  a[8]  = (_Float16)r2.x; a[9]  = (_Float16)r2.y; a[10] = (_Float16)r2.z; a[11] = (_Float16)r2.w;
  a[12] = (_Float16)r3.x; a[13] = (_Float16)r3.y; a[14] = (_Float16)r3.z; a[15] = (_Float16)r3.w;
  return a;
}

// B: 32x16 f16 tile where memory holds W as (N rows, 64 K cols) row-major,
// i.e. element (k,n) = W[n*64 + k].  Per-lane 16 contiguous floats.
// Contiguous k order used here (not ISA table order): lane n = lane&15; lanes 0-15 hold K k0..k0+15, lanes 16-31
// hold K k0+16..k0+31, half j -> k = k0 + hi*16 + j.
__device__ __forceinline__ v16h load_b_w64(const float* __restrict__ W,
                                           int k0, int n0, int lane) {
  const int ni = lane & 15, hi = lane >> 4;
  const float* p = W + (size_t)(n0 + ni) * KPAD + k0 + hi * 16;
  const float4 r0 = ((const float4*)p)[0];
  const float4 r1 = ((const float4*)p)[1];
  const float4 r2 = ((const float4*)p)[2];
  const float4 r3 = ((const float4*)p)[3];
  v16h b;
  b[0]  = (_Float16)r0.x; b[1]  = (_Float16)r0.y; b[2]  = (_Float16)r0.z; b[3]  = (_Float16)r0.w;
  b[4]  = (_Float16)r1.x; b[5]  = (_Float16)r1.y; b[6]  = (_Float16)r1.z; b[7]  = (_Float16)r1.w;
  b[8]  = (_Float16)r2.x; b[9]  = (_Float16)r2.y; b[10] = (_Float16)r2.z; b[11] = (_Float16)r2.w;
  b[12] = (_Float16)r3.x; b[13] = (_Float16)r3.y; b[14] = (_Float16)r3.z; b[15] = (_Float16)r3.w;
  return b;
}

// B: 32x16 f16 tile of a row-major (64 x 4096) activation matrix
// (element (k,n) at B[k*4096 + n]); coalesced across lanes, branch-free.
__device__ __forceinline__ v16h load_b_act(const float* __restrict__ B,
                                           int k0, int n0, int lane) {
  const int ni = lane & 15, hi = lane >> 4;
  const float* p = B + (size_t)(k0 + hi * 16) * L_TOT + n0 + ni;
  v16h b;
#pragma unroll
  for (int j = 0; j < 16; ++j) b[j] = (_Float16)p[(size_t)j * L_TOT];
  return b;
}

// D: 16x16 f32, full tile, unconditional store.
// VGPR r holds row m0+r (lanes 0-15) / m0+r+8 (lanes 16-31).
__device__ __forceinline__ void store_d(float* __restrict__ D, int ldd,
                                        int m0, int n0, const v8f& acc, int lane) {
  const int ni = lane & 15, hi = lane >> 4;
  float* p = D + (size_t)(m0 + hi * 8) * ldd + n0 + ni;
#pragma unroll
  for (int r = 0; r < 8; ++r) p[(size_t)r * ldd] = acc[r];
}

__device__ __forceinline__ float silu_f(float v) {
  return v / (1.0f + __expf(-v));
}
__device__ __forceinline__ float softplus_f(float v) {
  return (v > 20.0f) ? v : log1pf(__expf(v));
}

// ----------------------- 0) zero-padded staging ----------------------------
// dst is (dst_rows, 64) row-major; src rows are grouped: dst row r maps to
// src row (r/group)*valid_rows + (r%group), valid iff (r%group) < valid_rows.
__global__ void k_pad64(const float* __restrict__ src, float* __restrict__ dst,
                        int dst_rows, int group, int valid_rows, int src_cols) {
  const int idx = blockIdx.x * blockDim.x + threadIdx.x;
  if (idx >= dst_rows * KPAD) return;
  const int r = idx >> 6, c = idx & 63;
  const int rg = r % group, kg = r / group;
  float v = 0.0f;
  if (rg < valid_rows && c < src_cols)
    v = src[((size_t)kg * valid_rows + rg) * src_cols + c];
  dst[idx] = v;
}

// ----------------------- 1) in_proj: xz[l,e] = x[l,:] . W[e,:] -------------
// M=4096, N=96, K=48 (padded 64).  One wave per 16x16 output tile.
__global__ void __launch_bounds__(32)
k_in_proj(const float* __restrict__ xpad, const float* __restrict__ Wpad,
          float* __restrict__ xz) {
  const int lane = threadIdx.x;
  const int n0 = (blockIdx.x % 6) * 16;
  const int m0 = (blockIdx.x / 6) * 16;
  v8f acc = {};
#pragma unroll
  for (int k0 = 0; k0 < KPAD; k0 += 32) {
    v16h a = load_a64(xpad, m0, k0, lane);
    v16h b = load_b_w64(Wpad, k0, n0, lane);
    acc = __builtin_amdgcn_wmma_f32_16x16x32_f16(false, a, false, b,
                                                 (short)0, acc, false, false);
  }
  store_d(xz, 2 * DIN, m0, n0, acc, lane);
}

// ----------------------- 2) depthwise 3x3x3 conv + SiLU --------------------
__global__ void k_conv_silu(const float* __restrict__ xz, const float* __restrict__ cw,
                            const float* __restrict__ cb, float* __restrict__ xc) {
  const int idx = blockIdx.x * blockDim.x + threadIdx.x;
  if (idx >= DIN * L_TOT) return;
  const int d = idx >> 12;
  const int s = idx & (L_TOT - 1);
  const int h = s >> 8, w = (s >> 4) & 15, dd = s & 15;
  float acc = cb[d];
#pragma unroll
  for (int i = 0; i < 3; ++i) {
    const int hh = h + i - 1;
    if ((unsigned)hh >= 16u) continue;
#pragma unroll
    for (int j = 0; j < 3; ++j) {
      const int ww = w + j - 1;
      if ((unsigned)ww >= 16u) continue;
#pragma unroll
      for (int m = 0; m < 3; ++m) {
        const int dm = dd + m - 1;
        if ((unsigned)dm >= 16u) continue;
        const int pos = hh * 256 + ww * 16 + dm;
        acc += cw[d * 27 + i * 9 + j * 3 + m] * xz[pos * (2 * DIN) + d];
      }
    }
  }
  xc[idx] = silu_f(acc);
}

// ----------------------- 3) build 8 scan orderings (K-padded) --------------
// xs layout: (K=8, 64, 4096); rows d>=48 are zero so x_proj B loads are clean.
__global__ void k_gather_dirs(const float* __restrict__ xc, float* __restrict__ xs) {
  const int idx = blockIdx.x * blockDim.x + threadIdx.x;
  if (idx >= KDIR * KPAD * L_TOT) return;
  const int l = idx & (L_TOT - 1);
  const int d = (idx >> 12) & 63;
  const int k = idx / (KPAD * L_TOT);
  if (d >= DIN) { xs[idx] = 0.0f; return; }
  const int lp = (k >= 4) ? (L_TOT - 1 - l) : l;
  const int a = lp >> 8, b = (lp >> 4) & 15, c = lp & 15;
  int h, w, dd;
  switch (k & 3) {
    case 0: h = a;  w = b;  dd = c; break;   // (h,w,dd)
    case 1: w = a;  h = b;  dd = c; break;   // (w,h,dd)
    case 2: dd = a; w = b;  h = c;  break;   // (dd,w,h)
    default: h = a; dd = b; w = c;  break;   // (h,dd,w)
  }
  xs[idx] = xc[d * L_TOT + h * 256 + w * 16 + dd];
}

// ----------------------- 4) x_proj per direction (WMMA) --------------------
// xdbl[k][c][l] = sum_d xpw[k][c][d] * xs[k][d][l]; c tiles 3x16 (rows 35..47
// of the padded weight are zero, so full-tile stores are safe).
__global__ void __launch_bounds__(32)
k_x_proj(const float* __restrict__ xs, const float* __restrict__ xpw_pad,
         float* __restrict__ xdbl) {
  const int lane = threadIdx.x;
  int b = blockIdx.x;
  const int lt = b % 256; b /= 256;       // l tile (N dim)
  const int ct = b % 3;   b /= 3;         // c tile (M dim)
  const int k  = b;                        // direction
  const float* A  = xpw_pad + (size_t)k * DIN * KPAD;   // (48, 64) padded
  const float* Bm = xs      + (size_t)k * KPAD * L_TOT; // (64, 4096) padded
  float*       O  = xdbl    + (size_t)k * DIN * L_TOT;  // (48, 4096) padded
  const int m0 = ct * 16, n0 = lt * 16;
  v8f acc = {};
#pragma unroll
  for (int k0 = 0; k0 < KPAD; k0 += 32) {
    v16h a  = load_a64(A, m0, k0, lane);
    v16h bb = load_b_act(Bm, k0, n0, lane);
    acc = __builtin_amdgcn_wmma_f32_16x16x32_f16(false, a, false, bb,
                                                 (short)0, acc, false, false);
  }
  store_d(O, L_TOT, m0, n0, acc, lane);
}

// ----------------------- 5) selective scan ---------------------------------
// One wave per (k,d) channel.  Each lane owns a 128-element chunk.
__global__ void __launch_bounds__(32)
k_scan(const float* __restrict__ xs, const float* __restrict__ xdbl,
       const float* __restrict__ dtw, const float* __restrict__ dtb,
       const float* __restrict__ A_logs, const float* __restrict__ Dvec,
       float* __restrict__ y) {
  const int kd = blockIdx.x;            // 0..383 = k*48 + d
  const int k  = kd / DIN;
  const int d  = kd % DIN;
  const int lane = threadIdx.x;

  float Arow[NSTATE];
#pragma unroll
  for (int n = 0; n < NSTATE; ++n) Arow[n] = -__expf(A_logs[kd * NSTATE + n]);
  const float bias = dtb[kd];
  const float Dv   = Dvec[kd];
  const float w0 = dtw[kd * 3 + 0], w1 = dtw[kd * 3 + 1], w2 = dtw[kd * 3 + 2];

  const float* u   = xs + ((size_t)k * KPAD + d) * L_TOT;
  const float* xk  = xdbl + (size_t)k * DIN * L_TOT;
  const float* dt0 = xk;
  const float* dt1 = xk + L_TOT;
  const float* dt2 = xk + 2 * L_TOT;
  const float* Bmat = xk + 3 * L_TOT;
  const float* Cmat = xk + (3 + NSTATE) * L_TOT;
  float* yo = y + (size_t)kd * L_TOT;

  const int l0 = lane * 128;

  float aacc[NSTATE], bacc[NSTATE];
#pragma unroll
  for (int n = 0; n < NSTATE; ++n) { aacc[n] = 1.0f; bacc[n] = 0.0f; }

  for (int t = 0; t < 128; ++t) {
    const int l = l0 + t;
    __builtin_prefetch(u + l + 16, 0, 1);     // global_prefetch_b8
    const float uu    = u[l];
    const float delta = softplus_f(w0 * dt0[l] + w1 * dt1[l] + w2 * dt2[l] + bias);
    const float du    = delta * uu;
#pragma unroll
    for (int n = 0; n < NSTATE; ++n) {
      const float dA = __expf(delta * Arow[n]);
      const float Bv = Bmat[n * L_TOT + l];
      bacc[n] = dA * bacc[n] + du * Bv;
      aacc[n] *= dA;
    }
  }

  // Inclusive cross-lane scan of (a,b), (a1,b1)*(a2,b2) = (a2*a1, a2*b1+b2),
  // then shift to the exclusive carry-in for this lane's chunk.
#pragma unroll
  for (int n = 0; n < NSTATE; ++n) {
    float a = aacc[n], bsum = bacc[n];
    for (int off = 1; off < 32; off <<= 1) {
      const float ap = __shfl_up(a, off, 32);
      const float bp = __shfl_up(bsum, off, 32);
      if (lane >= off) { bsum = a * bp + bsum; a = a * ap; }
    }
    const float carry = __shfl_up(bsum, 1, 32);
    bacc[n] = (lane == 0) ? 0.0f : carry;   // reuse as h initial state
  }

  float h[NSTATE];
#pragma unroll
  for (int n = 0; n < NSTATE; ++n) h[n] = bacc[n];

  for (int t = 0; t < 128; ++t) {
    const int l = l0 + t;
    __builtin_prefetch(Cmat + l + 16, 0, 1);  // global_prefetch_b8
    const float uu    = u[l];
    const float delta = softplus_f(w0 * dt0[l] + w1 * dt1[l] + w2 * dt2[l] + bias);
    const float du    = delta * uu;
    float yl = Dv * uu;
#pragma unroll
    for (int n = 0; n < NSTATE; ++n) {
      const float dA = __expf(delta * Arow[n]);
      const float Bv = Bmat[n * L_TOT + l];
      h[n] = dA * h[n] + du * Bv;
      yl += h[n] * Cmat[n * L_TOT + l];
    }
    yo[l] = yl;
  }
}

// ----------------------- 6) merge directions -------------------------------
__global__ void k_combine(const float* __restrict__ y, float* __restrict__ ysum) {
  const int idx = blockIdx.x * blockDim.x + threadIdx.x;
  if (idx >= DIN * L_TOT) return;
  const int d = idx >> 12;
  const int pos = idx & (L_TOT - 1);
  const int h = pos >> 8, w = (pos >> 4) & 15, dd = pos & 15;
  const int l0 = pos;
  const int l1 = w * 256 + h * 16 + dd;
  const int l2 = dd * 256 + w * 16 + h;
  const int l3 = h * 256 + dd * 16 + w;
  float s = 0.0f;
  s += y[(0 * DIN + d) * L_TOT + l0] + y[(4 * DIN + d) * L_TOT + (L_TOT - 1 - l0)];
  s += y[(1 * DIN + d) * L_TOT + l1] + y[(5 * DIN + d) * L_TOT + (L_TOT - 1 - l1)];
  s += y[(2 * DIN + d) * L_TOT + l2] + y[(6 * DIN + d) * L_TOT + (L_TOT - 1 - l2)];
  s += y[(3 * DIN + d) * L_TOT + l3] + y[(7 * DIN + d) * L_TOT + (L_TOT - 1 - l3)];
  ysum[pos * DIN + d] = s;
}

// ----------------------- 7) LayerNorm + SiLU gate --------------------------
// Writes g as (4096, 64) row-major with zero pad so out_proj A loads are clean.
__global__ void k_ln_gate(const float* __restrict__ ysum, const float* __restrict__ xz,
                          const float* __restrict__ nw, const float* __restrict__ nb,
                          float* __restrict__ g) {
  const int pos = blockIdx.x * blockDim.x + threadIdx.x;
  if (pos >= L_TOT) return;
  const float* row = ysum + pos * DIN;
  float mu = 0.0f;
#pragma unroll 8
  for (int d = 0; d < DIN; ++d) mu += row[d];
  mu *= (1.0f / DIN);
  float s2 = 0.0f;
#pragma unroll 8
  for (int d = 0; d < DIN; ++d) { const float t = row[d] - mu; s2 += t * t; }
  const float inv = rsqrtf(s2 * (1.0f / DIN) + 1e-5f);
  float* go = g + (size_t)pos * KPAD;
#pragma unroll 8
  for (int d = 0; d < DIN; ++d) {
    const float zv = xz[pos * (2 * DIN) + DIN + d];
    go[d] = ((row[d] - mu) * inv * nw[d] + nb[d]) * silu_f(zv);
  }
#pragma unroll
  for (int d = DIN; d < KPAD; ++d) go[d] = 0.0f;
}

// ----------------------- 8) out_proj (WMMA) --------------------------------
// out[pos,c] = sum_e g[pos,e] * W[c,e].  M=4096, N=48, K=48 (padded 64).
__global__ void __launch_bounds__(32)
k_out_proj(const float* __restrict__ g, const float* __restrict__ Wpad,
           float* __restrict__ out) {
  const int lane = threadIdx.x;
  const int n0 = (blockIdx.x % 3) * 16;
  const int m0 = (blockIdx.x / 3) * 16;
  v8f acc = {};
#pragma unroll
  for (int k0 = 0; k0 < KPAD; k0 += 32) {
    v16h a = load_a64(g, m0, k0, lane);
    v16h b = load_b_w64(Wpad, k0, n0, lane);
    acc = __builtin_amdgcn_wmma_f32_16x16x32_f16(false, a, false, b,
                                                 (short)0, acc, false, false);
  }
  store_d(out, DIN, m0, n0, acc, lane);
}

// ---------------------------------------------------------------------------
extern "C" void kernel_launch(void* const* d_in, const int* in_sizes, int n_in,
                              void* d_out, int out_size, void* d_ws, size_t ws_size,
                              hipStream_t stream) {
  (void)in_sizes; (void)n_in; (void)out_size; (void)ws_size;
  const float* x    = (const float*)d_in[0];   // (4096, 48)
  const float* ipw  = (const float*)d_in[1];   // (96, 48)
  const float* cw   = (const float*)d_in[2];   // (48, 27)
  const float* cb   = (const float*)d_in[3];   // (48,)
  const float* xpw  = (const float*)d_in[4];   // (8, 35, 48)
  const float* dtw  = (const float*)d_in[5];   // (8, 48, 3)
  const float* dtb  = (const float*)d_in[6];   // (8, 48)
  const float* alog = (const float*)d_in[7];   // (384, 16)
  const float* Dv   = (const float*)d_in[8];   // (384,)
  const float* nw   = (const float*)d_in[9];   // (48,)
  const float* nb   = (const float*)d_in[10];  // (48,)
  const float* opw  = (const float*)d_in[11];  // (48, 48)

  float* ws = (float*)d_ws;
  float* xpad    = ws; ws += (size_t)L_TOT * KPAD;          // 262144
  float* Wpad    = ws; ws += (size_t)96 * KPAD;             //   6144
  float* xpwpad  = ws; ws += (size_t)KDIR * DIN * KPAD;     //  24576
  float* opwpad  = ws; ws += (size_t)DIN * KPAD;            //   3072
  float* xz      = ws; ws += (size_t)L_TOT * 2 * DIN;       // 393216
  float* xc      = ws; ws += (size_t)DIN * L_TOT;           // 196608
  float* xs      = ws; ws += (size_t)KDIR * KPAD * L_TOT;   // 2097152
  float* xdbl    = ws; ws += (size_t)KDIR * DIN * L_TOT;    // 1572864
  float* yb      = ws; ws += (size_t)KDIR * DIN * L_TOT;    // 1572864
  float* ysum    = ws; ws += (size_t)L_TOT * DIN;           // 196608
  float* g       = ws; ws += (size_t)L_TOT * KPAD;          // 262144
  float* out  = (float*)d_out;

  // Stage zero-padded operands.
  k_pad64<<<(L_TOT * KPAD + 255) / 256, 256, 0, stream>>>(x,   xpad,   L_TOT, L_TOT, L_TOT, DIN);
  k_pad64<<<(96 * KPAD + 255) / 256,    256, 0, stream>>>(ipw, Wpad,   96,    96,    96,    DIN);
  k_pad64<<<(KDIR * DIN * KPAD + 255) / 256, 256, 0, stream>>>(xpw, xpwpad, KDIR * DIN, DIN, CROWS, DIN);
  k_pad64<<<(DIN * KPAD + 255) / 256,   256, 0, stream>>>(opw, opwpad, DIN,   DIN,   DIN,   DIN);

  k_in_proj    <<<256 * 6, 32, 0, stream>>>(xpad, Wpad, xz);
  k_conv_silu  <<<(DIN * L_TOT + 255) / 256, 256, 0, stream>>>(xz, cw, cb, xc);
  k_gather_dirs<<<(KDIR * KPAD * L_TOT + 255) / 256, 256, 0, stream>>>(xc, xs);
  k_x_proj     <<<KDIR * 3 * 256, 32, 0, stream>>>(xs, xpwpad, xdbl);
  k_scan       <<<KDIR * DIN, 32, 0, stream>>>(xs, xdbl, dtw, dtb, alog, Dv, yb);
  k_combine    <<<(DIN * L_TOT + 255) / 256, 256, 0, stream>>>(yb, ysum);
  k_ln_gate    <<<(L_TOT + 63) / 64, 64, 0, stream>>>(ysum, xz, nw, nb, g);
  k_out_proj   <<<256 * 3, 32, 0, stream>>>(g, opwpad, out);
}
